// GCNN_attention_12086037971367
// MI455X (gfx1250) — matrix operand order repaired and checked by reading, compiled, not run on hardware
//
#include <hip/hip_runtime.h>
#include <hip/hip_bf16.h>
#include <math.h>

typedef __attribute__((ext_vector_type(16))) __bf16 v16bf;
typedef __attribute__((ext_vector_type(8)))  __bf16 v8bf;
typedef __attribute__((ext_vector_type(8)))  float  v8f;

#if defined(__has_builtin)
#  if __has_builtin(__builtin_amdgcn_tensor_load_to_lds) && __has_builtin(__builtin_amdgcn_s_wait_tensorcnt)
#    define HAVE_TDM 1
#  endif
#endif
#ifndef HAVE_TDM
#  define HAVE_TDM 0
#endif

namespace {
constexpr int cB   = 32;
constexpr int cNPG = 512;
constexpr int cN   = cB * cNPG;     // 16384
constexpr int cE   = 65536;
constexpr int cEC  = cE + cN;       // 81920 edges incl. self loops
constexpr int cF   = 128;
constexpr int cD   = 1024;
constexpr int cH   = 3;
constexpr int cHD  = cH * cD;       // 3072
constexpr int cK1 = 359, cK2 = 252, cK3 = 177;
constexpr int cLDT = 40;            // LDS row stride (ushorts) = 80B, 16B-aligned
constexpr float cNEG    = -1e9f;
constexpr float cNEGBIG = -1e30f;
}

// ---------------- fp32 -> bf16 hi/lo split (done ONCE per operand, not per tile) ------
__device__ __forceinline__ void split_bits(float x, unsigned short& hi, unsigned short& lo) {
  unsigned u = __float_as_uint(x) & 0xFFFF0000u;       // truncated hi -> exact remainder
  float fl = x - __uint_as_float(u);
  unsigned v = __float_as_uint(fl);
  v += 0x7FFFu + ((v >> 16) & 1u);                     // RNE for the lo plane
  hi = (unsigned short)(u >> 16);
  lo = (unsigned short)(v >> 16);
}

__global__ void convert_split_kernel(const float* __restrict__ src,
                                     unsigned short* __restrict__ hi,
                                     unsigned short* __restrict__ lo, size_t n) {
  size_t i = (size_t)blockIdx.x * blockDim.x + threadIdx.x;
  if (i >= n) return;
  unsigned short h, l;
  split_bits(src[i], h, l);
  hi[i] = h; lo[i] = l;
}

// tiled transpose: B[KxNc] fp32 -> BtHi/BtLo bf16 planes in [Nc][K] order
__global__ __launch_bounds__(256) void convert_splitT_kernel(
    const float* __restrict__ src, unsigned short* __restrict__ hiT,
    unsigned short* __restrict__ loT, int K, int Nc) {
  __shared__ float tile[32][33];
  const int k0 = blockIdx.y << 5, n0 = blockIdx.x << 5;
  const int t = threadIdx.x;
  const int tr = t >> 5, tc = t & 31;
  #pragma unroll
  for (int i = 0; i < 4; ++i)
    tile[tr + i * 8][tc] = src[(size_t)(k0 + tr + i * 8) * Nc + n0 + tc];
  __syncthreads();
  #pragma unroll
  for (int i = 0; i < 4; ++i) {
    int n = tr + i * 8;
    unsigned short h, l;
    split_bits(tile[tc][n], h, l);
    size_t o = (size_t)(n0 + n) * K + k0 + tc;
    hiT[o] = h; loT[o] = l;
  }
}

// ---------------- WMMA GEMM on preconverted bf16 planes -------------------------------
__device__ __forceinline__ v16bf mk16(v8bf a, v8bf b) {
  return __builtin_shufflevector(a, b, 0, 1, 2, 3, 4, 5, 6, 7, 8, 9, 10, 11, 12, 13, 14, 15);
}

// one K-step of fragment loads + 24 WMMAs (wave tile 32x64 within 128x128 block tile)
__device__ __forceinline__ void compute_step(
    const unsigned short* __restrict__ AsH, const unsigned short* __restrict__ AsL,
    const unsigned short* __restrict__ BsH, const unsigned short* __restrict__ BsL,
    int waveM, int waveN, int halfsel, int l15, v8f (&acc)[2][4])
{
  v16bf aH[2], aL[2];
  #pragma unroll
  for (int mt = 0; mt < 2; ++mt) {
    const int r = (waveM << 5) + (mt << 4) + l15;
    const unsigned short* p = &AsH[r * cLDT + (halfsel << 3)];
    const unsigned short* q = &AsL[r * cLDT + (halfsel << 3)];
    aH[mt] = mk16(*(const v8bf*)p, *(const v8bf*)(p + 16));
    aL[mt] = mk16(*(const v8bf*)q, *(const v8bf*)(q + 16));
  }
  #pragma unroll
  for (int j = 0; j < 4; ++j) {
    const int nr = (waveN << 6) + (j << 4) + l15;
    const unsigned short* p = &BsH[nr * cLDT + (halfsel << 3)];
    const unsigned short* q = &BsL[nr * cLDT + (halfsel << 3)];
    const v16bf bH = mk16(*(const v8bf*)p, *(const v8bf*)(p + 16));
    const v16bf bL = mk16(*(const v8bf*)q, *(const v8bf*)(q + 16));
    #pragma unroll
    for (int mt = 0; mt < 2; ++mt) {
      acc[mt][j] = __builtin_amdgcn_wmma_f32_16x16x32_bf16(false, aH[mt], false, bH, (short)0, acc[mt][j], false, false);
      acc[mt][j] = __builtin_amdgcn_wmma_f32_16x16x32_bf16(false, aH[mt], false, bL, (short)0, acc[mt][j], false, false);
      acc[mt][j] = __builtin_amdgcn_wmma_f32_16x16x32_bf16(false, aL[mt], false, bH, (short)0, acc[mt][j], false, false);
    }
  }
}

#if HAVE_TDM
typedef __attribute__((ext_vector_type(4))) unsigned int u32x4;
typedef __attribute__((ext_vector_type(8))) int i32x8;
typedef __attribute__((ext_vector_type(4))) int i32x4;

__device__ __forceinline__ unsigned lds_byte_off(const void* p) {
  return (unsigned)(size_t)(const __attribute__((address_space(3))) char*)p;
}

// Tensor DMA: 2-D tile 32(K) x 128(rows) of 2-byte data, global->LDS, with LDS row
// padding 4 DWORDs per 16 DWORDs (64B row -> 80B stride), OOB rows return zero.
__device__ __forceinline__ void tdm_load_tile2d(
    const unsigned short* gbase, unsigned ldsOff,
    unsigned rowsRemain, unsigned colsRemain, unsigned strideElems)
{
  unsigned long long ga = (unsigned long long)(size_t)gbase;
  u32x4 g0;
  g0.x = 1u;                                                  // count=1, user descriptor
  g0.y = ldsOff;                                              // lds_addr (bytes)
  g0.z = (unsigned)ga;                                        // global_addr[31:0]
  g0.w = (unsigned)((ga >> 32) & 0x01FFFFFFu) | (2u << 30);   // addr[56:32] | type=2
  i32x8 g1;
  g1[0] = (1 << 16)      // data_size = 2 bytes
        | (1 << 20)      // pad_enable
        | (3 << 22)      // pad_interval: every 16 DWORDs (64B)
        | (3 << 25);     // pad_amount: 4 DWORDs (16B)
  g1[1] = (int)((colsRemain & 0xFFFFu) << 16);                           // tensor_dim0 lo
  g1[2] = (int)(((colsRemain >> 16) & 0xFFFFu) | ((rowsRemain & 0xFFFFu) << 16));
  g1[3] = (int)(((rowsRemain >> 16) & 0xFFFFu) | (32u << 16));           // tile_dim0=32
  g1[4] = 128;                                                           // tile_dim1=128
  g1[5] = (int)strideElems;                                              // dim0 stride lo32
  g1[6] = 0;
  g1[7] = 0;
  i32x4 g2 = {0, 0, 0, 0};
  i32x4 g3 = {0, 0, 0, 0};
#if __clang_major__ >= 23
  i32x8 g4 = {0, 0, 0, 0, 0, 0, 0, 0};
  __builtin_amdgcn_tensor_load_to_lds(g0, g1, g2, g3, g4, 0);
#else
  __builtin_amdgcn_tensor_load_to_lds(g0, g1, g2, g3, 0);
#endif
}
#endif  // HAVE_TDM

// C[M,Nc] = A[M,K] @ B[K,Nc] (+bias)(+relu), split-bf16 3-product f32 accumulation.
// Requirements: K % 32 == 0, Nc % 128 == 0 (M guarded).
// Block tile 128x128: 8 wave32s as 4(M) x 2(N); double-buffered LDS; staging via TDM
// (wave 0 issues 4 tensor_load_to_lds per K-step) or manual uint4 copies as fallback.
__global__ __launch_bounds__(256) void gemm_planes_wmma(
    const unsigned short* __restrict__ Ahi, const unsigned short* __restrict__ Alo,
    const unsigned short* __restrict__ BtHi, const unsigned short* __restrict__ BtLo,
    const float* __restrict__ bias, float* __restrict__ C,
    int M, int K, int Nc, int relu)
{
  __shared__ unsigned short AsH[2][128 * cLDT];
  __shared__ unsigned short AsL[2][128 * cLDT];
  __shared__ unsigned short BsH[2][128 * cLDT];
  __shared__ unsigned short BsL[2][128 * cLDT];

  const int t = threadIdx.x;
  const int lane = t & 31;
  const int wave = t >> 5;
  const int waveM = wave & 3;             // *32 -> M offset in block tile
  const int waveN = wave >> 2;            // *64 -> N offset in block tile
  const int halfsel = lane >> 4;
  const int l15 = lane & 15;
  const int m0 = blockIdx.y << 7;
  const int n0 = blockIdx.x << 7;
  const int nk = K >> 5;

  v8f acc[2][4] = {};

#if HAVE_TDM
  if (wave == 0) {
    tdm_load_tile2d(Ahi  + (size_t)m0 * K, lds_byte_off(AsH[0]), M - m0, K, K);
    tdm_load_tile2d(Alo  + (size_t)m0 * K, lds_byte_off(AsL[0]), M - m0, K, K);
    tdm_load_tile2d(BtHi + (size_t)n0 * K, lds_byte_off(BsH[0]), Nc - n0, K, K);
    tdm_load_tile2d(BtLo + (size_t)n0 * K, lds_byte_off(BsL[0]), Nc - n0, K, K);
  }
  for (int kt = 0; kt < nk; ++kt) {
    const int cur = kt & 1;
    if (wave == 0) {
      if (kt + 1 < nk) {
        const int nxt = cur ^ 1;
        const int kc = (kt + 1) << 5;
        tdm_load_tile2d(Ahi  + (size_t)m0 * K + kc, lds_byte_off(AsH[nxt]), M - m0, K - kc, K);
        tdm_load_tile2d(Alo  + (size_t)m0 * K + kc, lds_byte_off(AsL[nxt]), M - m0, K - kc, K);
        tdm_load_tile2d(BtHi + (size_t)n0 * K + kc, lds_byte_off(BsH[nxt]), Nc - n0, K - kc, K);
        tdm_load_tile2d(BtLo + (size_t)n0 * K + kc, lds_byte_off(BsL[nxt]), Nc - n0, K - kc, K);
        __builtin_amdgcn_s_wait_tensorcnt(4);   // current batch complete, next in flight
      } else {
        __builtin_amdgcn_s_wait_tensorcnt(0);
      }
    }
    __syncthreads();   // buffer[cur] visible to all waves
    compute_step(AsH[cur], AsL[cur], BsH[cur], BsL[cur], waveM, waveN, halfsel, l15, acc);
    __syncthreads();   // all done reading buffer[cur] before it is refilled
  }
#else
  const int srow = t >> 1;                // staging: 2 threads per 32-wide row
  const int skc  = (t & 1) << 4;
  for (int kt = 0; kt < nk; ++kt) {
    const int cur = kt & 1;
    const int k0 = kt << 5;
    uint4 a0 = {}, a1 = {}, a2 = {}, a3 = {};
    const int gm = m0 + srow;
    if (gm < M) {
      const size_t ga = (size_t)gm * K + k0 + skc;
      a0 = *(const uint4*)(Ahi + ga);
      a1 = *(const uint4*)(Ahi + ga + 8);
      a2 = *(const uint4*)(Alo + ga);
      a3 = *(const uint4*)(Alo + ga + 8);
    }
    const size_t gb = (size_t)(n0 + srow) * K + k0 + skc;
    const uint4 b0 = *(const uint4*)(BtHi + gb);
    const uint4 b1 = *(const uint4*)(BtHi + gb + 8);
    const uint4 b2 = *(const uint4*)(BtLo + gb);
    const uint4 b3 = *(const uint4*)(BtLo + gb + 8);
    __syncthreads();
    *(uint4*)&AsH[cur][srow * cLDT + skc]     = a0;
    *(uint4*)&AsH[cur][srow * cLDT + skc + 8] = a1;
    *(uint4*)&AsL[cur][srow * cLDT + skc]     = a2;
    *(uint4*)&AsL[cur][srow * cLDT + skc + 8] = a3;
    *(uint4*)&BsH[cur][srow * cLDT + skc]     = b0;
    *(uint4*)&BsH[cur][srow * cLDT + skc + 8] = b1;
    *(uint4*)&BsL[cur][srow * cLDT + skc]     = b2;
    *(uint4*)&BsL[cur][srow * cLDT + skc + 8] = b3;
    __syncthreads();
    compute_step(AsH[cur], AsL[cur], BsH[cur], BsL[cur], waveM, waveN, halfsel, l15, acc);
  }
#endif

  // C/D layout: VGPR r -> M=r (lanes 0-15) / M=8+r (lanes 16-31), N = lane%16
  #pragma unroll
  for (int mt = 0; mt < 2; ++mt) {
    #pragma unroll
    for (int j = 0; j < 4; ++j) {
      const int nc = n0 + (waveN << 6) + (j << 4) + l15;
      #pragma unroll
      for (int r = 0; r < 8; ++r) {
        const int gm = m0 + (waveM << 5) + (mt << 4) + r + (halfsel << 3);
        if (gm < M) {
          float v = acc[mt][j][r];
          if (bias) v += bias[nc];
          if (relu) v = fmaxf(v, 0.0f);
          C[(size_t)gm * Nc + nc] = v;
        }
      }
    }
  }
}

// ---------------- graph / attention kernels ----------------
__global__ void mask_init_kernel(int* nmask, int* evalid) {
  int i = blockIdx.x * blockDim.x + threadIdx.x;
  if (i < cN) nmask[i] = 1;
  if (i < cE) evalid[i] = 1;
}

__global__ void attn_init_kernel(float* __restrict__ gat, const float* __restrict__ b,
                                 float* __restrict__ maxbuf, float* __restrict__ den) {
  int i = blockIdx.x * blockDim.x + threadIdx.x;
  if (i < cN * cHD) gat[i] = b[i % cHD];
  if (i < cN * cH) { maxbuf[i] = cNEG; den[i] = 0.0f; }
}

__global__ __launch_bounds__(256) void node_dots_kernel(
    const float* __restrict__ xs, const float* __restrict__ a_s,
    const float* __restrict__ a_d, float* __restrict__ asum, float* __restrict__ adsum) {
  __shared__ float r1[256], r2[256];
  int n = blockIdx.x, t = threadIdx.x;
  const float* row = xs + (size_t)n * cHD;
  for (int hh = 0; hh < cH; ++hh) {
    float s1 = 0.0f, s2 = 0.0f;
    for (int d = t; d < cD; d += 256) {
      float v = row[hh * cD + d];
      s1 += v * a_s[hh * cD + d];
      s2 += v * a_d[hh * cD + d];
    }
    r1[t] = s1; r2[t] = s2; __syncthreads();
    for (int st = 128; st > 0; st >>= 1) {
      if (t < st) { r1[t] += r1[t + st]; r2[t] += r2[t + st]; }
      __syncthreads();
    }
    if (t == 0) { asum[n * cH + hh] = r1[0]; adsum[n * cH + hh] = r2[0]; }
    __syncthreads();
  }
}

__device__ __forceinline__ void edge_sd(const int* src0, const int* dst0,
                                        const int* evalid, const int* nmask,
                                        int e, int& s, int& d, bool& m) {
  if (e < cE) { s = src0[e]; d = dst0[e]; m = evalid[e] != 0; }
  else        { s = d = e - cE; m = nmask[s] != 0; }
}
__device__ __forceinline__ float edge_logit(const float* asum, const float* adsum,
                                            int s, int d, int hh, bool m) {
  float l = asum[s * cH + hh] + adsum[d * cH + hh];
  l = (l >= 0.0f) ? l : 0.2f * l;            // leaky_relu slope 0.2
  return m ? l : cNEG;
}
__device__ __forceinline__ void atomicMaxF(float* addr, float val) {
  if (val >= 0.0f) atomicMax((int*)addr, __float_as_int(val));
  else             atomicMin((unsigned int*)addr, (unsigned int)__float_as_int(val));
}

__global__ void attn_pass1(const int* src0, const int* dst0, const int* evalid,
                           const int* nmask, const float* asum, const float* adsum,
                           float* maxbuf) {
  int i = blockIdx.x * blockDim.x + threadIdx.x;
  if (i >= cEC * cH) return;
  int e = i / cH, hh = i % cH, s, d; bool m;
  edge_sd(src0, dst0, evalid, nmask, e, s, d, m);
  atomicMaxF(&maxbuf[d * cH + hh], edge_logit(asum, adsum, s, d, hh, m));
}

__global__ void attn_pass2(const int* src0, const int* dst0, const int* evalid,
                           const int* nmask, const float* asum, const float* adsum,
                           const float* maxbuf, float* den, float* exbuf) {
  int i = blockIdx.x * blockDim.x + threadIdx.x;
  if (i >= cEC * cH) return;
  int e = i / cH, hh = i % cH, s, d; bool m;
  edge_sd(src0, dst0, evalid, nmask, e, s, d, m);
  float ex = expf(edge_logit(asum, adsum, s, d, hh, m) - maxbuf[d * cH + hh]);
  exbuf[i] = ex;
  atomicAdd(&den[d * cH + hh], ex);
}

__global__ __launch_bounds__(256) void attn_pass3(
    const int* src0, const int* dst0, const float* __restrict__ exbuf,
    const float* __restrict__ den, const float* __restrict__ xs, float* __restrict__ gat) {
  int e = blockIdx.x, hh = blockIdx.y, s, d;
  if (e < cE) { s = src0[e]; d = dst0[e]; }
  else        { s = d = e - cE; }
  float ex = exbuf[e * cH + hh];
  if (ex == 0.0f) return;
  float alpha = ex / den[d * cH + hh];
  const float* srow = xs + (size_t)s * cHD + hh * cD;
  float* drow = gat + (size_t)d * cHD + hh * cD;
  for (int dd = threadIdx.x; dd < cD; dd += 256)
    atomicAdd(&drow[dd], alpha * srow[dd]);
}

__global__ void pnorm_kernel(const float* __restrict__ p, float* invn) {
  __shared__ float r[256];
  int t = threadIdx.x;
  float s = 0.0f;
  for (int d = t; d < cD; d += 256) s += p[d] * p[d];
  r[t] = s; __syncthreads();
  for (int st = 128; st > 0; st >>= 1) { if (t < st) r[t] += r[t + st]; __syncthreads(); }
  if (t == 0) invn[0] = 1.0f / sqrtf(r[0]);
}

__global__ __launch_bounds__(256) void score_kernel(
    const float* __restrict__ h, const float* __restrict__ p,
    const float* __restrict__ invn, float* __restrict__ score) {
  __shared__ float r[256];
  int n = blockIdx.x, t = threadIdx.x;
  const float* row = h + (size_t)n * cD;
  float s = 0.0f;
  for (int d = t; d < cD; d += 256) s += row[d] * p[d];
  r[t] = s; __syncthreads();
  for (int st = 128; st > 0; st >>= 1) { if (t < st) r[t] += r[t + st]; __syncthreads(); }
  if (t == 0) score[n] = r[0] * invn[0];
}

__global__ __launch_bounds__(256) void topk_kernel(
    const float* __restrict__ score, const int* __restrict__ nm_in,
    int* __restrict__ nm_out, int kkeep) {
  __shared__ float skey[cNPG];
  __shared__ int   sidx[cNPG];
  int b = blockIdx.x, t = threadIdx.x;
  for (int i = t; i < cNPG; i += 256) {
    int n = b * cNPG + i;
    skey[i] = nm_in[n] ? score[n] : cNEGBIG;
    sidx[i] = i;
  }
  __syncthreads();
  for (int ksz = 2; ksz <= cNPG; ksz <<= 1) {
    for (int j = ksz >> 1; j > 0; j >>= 1) {
      for (int i = t; i < cNPG; i += 256) {
        int pr = i ^ j;
        if (pr > i) {
          bool desc = ((i & ksz) == 0);
          bool sw = desc ? (skey[i] < skey[pr]) : (skey[i] > skey[pr]);
          if (sw) {
            float tk = skey[i]; skey[i] = skey[pr]; skey[pr] = tk;
            int ti = sidx[i]; sidx[i] = sidx[pr]; sidx[pr] = ti;
          }
        }
      }
      __syncthreads();
    }
  }
  for (int i = t; i < cNPG; i += 256)
    nm_out[b * cNPG + sidx[i]] = (i < kkeep) ? 1 : 0;
}

__global__ void gate_kernel(float* __restrict__ h, const float* __restrict__ score,
                            const int* __restrict__ nm) {
  int i = blockIdx.x * blockDim.x + threadIdx.x;
  if (i >= cN * cD) return;
  int n = i / cD;
  h[i] = nm[n] ? h[i] * tanhf(score[n]) : 0.0f;
}

__global__ void evalid_kernel(int* __restrict__ evalid, const int* src0,
                              const int* dst0, const int* __restrict__ nm) {
  int e = blockIdx.x * blockDim.x + threadIdx.x;
  if (e >= cE) return;
  evalid[e] = (evalid[e] && nm[src0[e]] && nm[dst0[e]]) ? 1 : 0;
}

__global__ void gpool_kernel(const float* __restrict__ h, const int* __restrict__ nm,
                             float kdiv, float* __restrict__ zbuf, int first) {
  int t = blockIdx.x * blockDim.x + threadIdx.x;
  if (t >= cB * cD) return;
  int b = t / cD, d = t % cD;
  float s = 0.0f, mx = cNEGBIG;
  const float* base = h + (size_t)b * cNPG * cD + d;
  for (int n = 0; n < cNPG; ++n) {
    if (nm[b * cNPG + n]) {
      float v = base[(size_t)n * cD];
      s += v; mx = fmaxf(mx, v);
    }
  }
  float mean = s / kdiv;
  if (first) { zbuf[b * 2 * cD + d] = mean;  zbuf[b * 2 * cD + cD + d] = mx; }
  else       { zbuf[b * 2 * cD + d] += mean; zbuf[b * 2 * cD + cD + d] += mx; }
}

__global__ void fc2_kernel(const float* __restrict__ zr, const float* __restrict__ w,
                           const float* __restrict__ bias, float* __restrict__ out) {
  int t = threadIdx.x;
  if (t >= cB * 2) return;
  int r = t >> 1, c = t & 1;
  float s = bias[c];
  for (int k = 0; k < cD; ++k) s += zr[(size_t)r * cD + k] * w[k * 2 + c];
  out[r * 2 + c] = s;
}

// ---------------- host orchestration ----------------
extern "C" void kernel_launch(void* const* d_in, const int* in_sizes, int n_in,
                              void* d_out, int out_size, void* d_ws, size_t ws_size,
                              hipStream_t stream) {
  (void)in_sizes; (void)n_in; (void)out_size; (void)ws_size;
  const float* x    = (const float*)d_in[0];
  const int*   eidx = (const int*)d_in[2];
  const int* src0 = eidx;
  const int* dst0 = eidx + cE;
  const float* W[3]   = {(const float*)d_in[4],  (const float*)d_in[11], (const float*)d_in[18]};
  const float* Asr[3] = {(const float*)d_in[5],  (const float*)d_in[12], (const float*)d_in[19]};
  const float* Adr[3] = {(const float*)d_in[6],  (const float*)d_in[13], (const float*)d_in[20]};
  const float* Bb[3]  = {(const float*)d_in[7],  (const float*)d_in[14], (const float*)d_in[21]};
  const float* Lw[3]  = {(const float*)d_in[8],  (const float*)d_in[15], (const float*)d_in[22]};
  const float* Lb[3]  = {(const float*)d_in[9],  (const float*)d_in[16], (const float*)d_in[23]};
  const float* Pp[3]  = {(const float*)d_in[10], (const float*)d_in[17], (const float*)d_in[24]};
  const float* f1w = (const float*)d_in[25];
  const float* f1b = (const float*)d_in[26];
  const float* f2w = (const float*)d_in[27];
  const float* f2b = (const float*)d_in[28];
  const int kkeep[3] = {cK1, cK2, cK3};

  float* ws = (float*)d_ws;
  size_t off = 0;
  auto alloc = [&](size_t n) {                 // 64B-aligned bump allocator
    n = (n + 15) & ~(size_t)15;
    float* p = ws + off; off += n; return p;
  };
  float* xs     = alloc((size_t)cN * cHD);     // GAT transformed features
  float* gat    = alloc((size_t)cN * cHD);     // attention output (+bias)
  float* h      = alloc((size_t)cN * cD);      // post-linear / gated features
  unsigned short* btHi = (unsigned short*)alloc((size_t)cHD * cHD / 2); // max K*Nc plane
  unsigned short* btLo = (unsigned short*)alloc((size_t)cHD * cHD / 2);
  float* asum   = alloc((size_t)cN * cH);
  float* adsum  = alloc((size_t)cN * cH);
  float* maxbuf = alloc((size_t)cN * cH);
  float* den    = alloc((size_t)cN * cH);
  float* exbuf  = alloc((size_t)cEC * cH);
  float* score  = alloc((size_t)cN);
  float* invn   = alloc(16);
  float* zbuf   = alloc((size_t)cB * 2 * cD);
  float* zr     = alloc((size_t)cB * cD);
  int* maskA  = (int*)alloc((size_t)cN);
  int* maskB  = (int*)alloc((size_t)cN);
  int* evalid = (int*)alloc((size_t)cE);

  // split-convert + transpose-convert + WMMA GEMM; A planes live in a dead fp32 region
  auto run_gemm = [&](const float* Afp, const float* Bfp, const float* bias, float* Cout,
                      int M, int K, int Nc, int relu, float* planeRegion) {
    unsigned short* aHi = (unsigned short*)planeRegion;
    unsigned short* aLo = aHi + (size_t)M * K;
    size_t nA = (size_t)M * K;
    convert_split_kernel<<<(unsigned)((nA + 255) / 256), 256, 0, stream>>>(Afp, aHi, aLo, nA);
    convert_splitT_kernel<<<dim3(Nc / 32, K / 32), 256, 0, stream>>>(Bfp, btHi, btLo, K, Nc);
    gemm_planes_wmma<<<dim3(Nc / 128, (M + 127) / 128), 256, 0, stream>>>(
        aHi, aLo, btHi, btLo, bias, Cout, M, K, Nc, relu);
  };

  mask_init_kernel<<<(cE + 255) / 256, 256, 0, stream>>>(maskA, evalid);

  int* nmIn = maskA;
  int* nmOut = maskB;
  const float* xin = x;
  int Kin = cF;
  for (int s = 0; s < 3; ++s) {
    // xs = xin @ W   (A planes staged in gat region: gat is re-initialized just after)
    run_gemm(xin, W[s], nullptr, xs, cN, Kin, cHD, 0, gat);
    node_dots_kernel<<<cN, 256, 0, stream>>>(xs, Asr[s], Adr[s], asum, adsum);
    attn_init_kernel<<<(unsigned)(((size_t)cN * cHD + 255) / 256), 256, 0, stream>>>(
        gat, Bb[s], maxbuf, den);
    attn_pass1<<<(cEC * cH + 255) / 256, 256, 0, stream>>>(
        src0, dst0, evalid, nmIn, asum, adsum, maxbuf);
    attn_pass2<<<(cEC * cH + 255) / 256, 256, 0, stream>>>(
        src0, dst0, evalid, nmIn, asum, adsum, maxbuf, den, exbuf);
    attn_pass3<<<dim3(cEC, cH), 256, 0, stream>>>(src0, dst0, exbuf, den, xs, gat);
    // h = gat @ lw + lb   (A planes staged in xs region: xs is dead after pass3)
    run_gemm(gat, Lw[s], Lb[s], h, cN, cHD, cD, 0, xs);
    pnorm_kernel<<<1, 256, 0, stream>>>(Pp[s], invn);
    score_kernel<<<cN, 256, 0, stream>>>(h, Pp[s], invn, score);
    topk_kernel<<<cB, 256, 0, stream>>>(score, nmIn, nmOut, kkeep[s]);
    gate_kernel<<<(unsigned)(((size_t)cN * cD + 255) / 256), 256, 0, stream>>>(
        h, score, nmOut);
    evalid_kernel<<<(cE + 255) / 256, 256, 0, stream>>>(evalid, src0, dst0, nmOut);
    gpool_kernel<<<(cB * cD + 255) / 256, 256, 0, stream>>>(
        h, nmOut, (float)kkeep[s], zbuf, s == 0 ? 1 : 0);
    int* tmp = nmIn; nmIn = nmOut; nmOut = tmp;
    xin = h; Kin = cD;
  }

  // zr = relu(z @ f1w + f1b)   (tiny GEMM, M=32; planes in gat region)
  run_gemm(zbuf, f1w, f1b, zr, cB, 2 * cD, cD, 1, gat);
  fc2_kernel<<<1, 64, 0, stream>>>(zr, f2w, f2b, (float*)d_out);
}
